// TransformerEncoderLayer_1_55241869361204
// MI455X (gfx1250) — matrix operand order repaired and checked by reading, not compiled, not run on hardware
//
#include <hip/hip_runtime.h>
#include <hip/hip_bf16.h>
#include <math.h>

// ---------------------------------------------------------------------------
// CDNA5 (gfx1250) wave32 WMMA transformer encoder layer.
// All GEMMs: D = A(16x32 bf16) x B(32x16 bf16) + C(16x16 f32) via
// v_wmma_f32_16x16x32_bf16. One wave computes a 16x64 output tile (4 accs).
// ---------------------------------------------------------------------------

#define DI __device__ __forceinline__

typedef __attribute__((ext_vector_type(16))) __bf16 bf16x16;
typedef __attribute__((ext_vector_type(8)))  __bf16 bf16x8;
typedef __attribute__((ext_vector_type(8)))  float  f32x8;

union BF16V {
  bf16x16 v;
  bf16x8  h[2];
  __bf16  e[16];
};

DI bf16x16 join8(bf16x8 lo, bf16x8 hi) {
  BF16V u; u.h[0] = lo; u.h[1] = hi; return u.v;
}

DI bf16x16 zero16() {
  BF16V u;
#pragma unroll
  for (int i = 0; i < 16; ++i) u.e[i] = (__bf16)0.0f;
  return u.v;
}

// A fragment (16x32 bf16, M x K): lane li = row; lanes 0-15 hold K = k0+[0..7]
// and k0+[16..23]; lanes 16-31 hold K = k0+[8..15] and k0+[24..31].
// rowbase points at the lane's row (K-contiguous, arbitrary row stride).
DI bf16x16 load_a(const __bf16* rowbase, int k0, int half) {
  const __bf16* p = rowbase + k0 + half * 8;
  return join8(*(const bf16x8*)p, *(const bf16x8*)(p + 16));
}

// B fragment (32x16 bf16, K x N): lane li = column; lanes 0-15 hold
// K = k0+[0..7] and k0+[16..23], lanes 16-31 hold K = k0+[8..15] and
// k0+[24..31]. colbase points at the lane's column storage, K-contiguous (stride 1 in K).
DI bf16x16 load_b(const __bf16* colbase, int k0, int half) {
  const __bf16* p = colbase + k0 + half * 8;
  return join8(*(const bf16x8*)p, *(const bf16x8*)(p + 16));
}

DI f32x8 wmma_bf16f32(bf16x16 a, bf16x16 b, f32x8 c) {
  return __builtin_amdgcn_wmma_f32_16x16x32_bf16(
      /*neg_a=*/false, a, /*neg_b=*/false, b,
      /*c_mod=*/(short)0, c, /*reuse_a=*/false, /*reuse_b=*/false);
}

// Problem constants
#define BATCH 2
#define CH    256
#define NPIX  2304   // 48*48
#define HEADS 8
#define HD    32
#define DCONV 1024

// ---------------------------------------------------------------------------
// Kernel 1: pixel norm (RMS over channels) + pack to bf16 pixel-major rows.
// in:  f32 [b][C][NPIX] (NCHW), alpha f32 [C]
// out: bf16 [b][NPIX][C]   (GEMM A layout, K = channels contiguous)
// ---------------------------------------------------------------------------
__global__ void pixelnorm_pack(const float* __restrict__ in,
                               const float* __restrict__ alpha,
                               __bf16* __restrict__ out) {
  int p = blockIdx.x % NPIX;
  int b = blockIdx.x / NPIX;
  int c = threadIdx.x;           // blockDim.x == 256
  float v = in[(b * CH + c) * NPIX + p];
  __shared__ float red[CH];
  red[c] = v * v;
  __syncthreads();
  for (int s = CH / 2; s > 0; s >>= 1) {
    if (c < s) red[c] += red[c + s];
    __syncthreads();
  }
  float rms = rsqrtf(red[0] * (1.0f / CH) + 1e-8f);
  out[(b * NPIX + p) * CH + c] = (__bf16)(alpha[c] * v * rms);
}

// ---------------------------------------------------------------------------
// Kernel 2: f32 -> bf16 copy (weight packing)
// ---------------------------------------------------------------------------
__global__ void f32_to_bf16(const float* __restrict__ in,
                            __bf16* __restrict__ out, int n) {
  int i = blockIdx.x * blockDim.x + threadIdx.x;
  if (i < n) out[i] = (__bf16)in[i];
}

// conv1_w [1024][256][3][3] f32  ->  w1p [9][1024][256] bf16 (B-fragment layout)
__global__ void pack_conv1(const float* __restrict__ w,
                           __bf16* __restrict__ out) {
  int i = blockIdx.x * blockDim.x + threadIdx.x;
  if (i >= DCONV * CH * 9) return;
  int t  = i % 9;
  int ic = (i / 9) % CH;
  int oc = i / (9 * CH);
  out[(t * DCONV + oc) * CH + ic] = (__bf16)w[i];
}

// ---------------------------------------------------------------------------
// Kernel 3: fused QKV projections.  q = xn @ wz^T + bz  (z = 0,1,2)
// M = NPIX rows (pixels), N = 256, K = 256.  Wave computes 16x64 tile.
// Q,K written bf16 [b][h][n][32]; V written TRANSPOSED bf16 [b][h][32][n].
// ---------------------------------------------------------------------------
__global__ void qkv_gemm(const __bf16* __restrict__ xn,
                         const __bf16* __restrict__ wqkv,   // [3][256][256]
                         const float* __restrict__ bq,
                         const float* __restrict__ bk,
                         const float* __restrict__ bv,
                         __bf16* __restrict__ q,
                         __bf16* __restrict__ kk,
                         __bf16* __restrict__ vT) {
  int id = blockIdx.x;
  int ng = id & 3;   id >>= 2;     // 4 groups of 64 output cols
  int mt = id % 144; id /= 144;    // 144 row tiles of 16
  int b  = id % BATCH;
  int z  = id / BATCH;             // 0=Q 1=K 2=V
  int lane = threadIdx.x, li = lane & 15, half = lane >> 4;

  const __bf16* A = xn + ((size_t)(b * NPIX + mt * 16 + li)) * CH;
  const __bf16* W = wqkv + (size_t)z * CH * CH;
  int n0 = ng * 64;

  f32x8 acc[4] = {};
  for (int k0 = 0; k0 < CH; k0 += 32) {
    bf16x16 a = load_a(A, k0, half);
#pragma unroll
    for (int nt = 0; nt < 4; ++nt) {
      bf16x16 bf = load_b(W + (size_t)(n0 + nt * 16 + li) * CH, k0, half);
      acc[nt] = wmma_bf16f32(a, bf, acc[nt]);
    }
  }

  const float* bias = (z == 0) ? bq : (z == 1) ? bk : bv;
#pragma unroll
  for (int nt = 0; nt < 4; ++nt) {
    int col = n0 + nt * 16 + li;       // output channel 0..255
    int h = col >> 5, d = col & 31;
    float bb = bias[col];
#pragma unroll
    for (int r = 0; r < 8; ++r) {
      int row = mt * 16 + half * 8 + r;    // C layout: M = half*8 + r
      float val = acc[nt][r] + bb;
      if (z == 2)
        vT[((size_t)(b * HEADS + h) * HD + d) * NPIX + row] = (__bf16)val;
      else {
        __bf16* dst = (z == 0) ? q : kk;
        dst[((size_t)(b * HEADS + h) * NPIX + row) * HD + d] = (__bf16)val;
      }
    }
  }
}

// ---------------------------------------------------------------------------
// Kernel 4: streaming-softmax (flash) attention.
// One wave per 16-row q-tile per (b,head). d = 32 => single K-step per WMMA.
// Loops over 32 key columns at a time: 2 WMMAs for S, online max/sum via
// 16-lane shfl_xor reductions, P through LDS (C-layout -> A-layout), 2 WMMAs
// accumulating O. Writes bf16 [b][n][256].
// ---------------------------------------------------------------------------
__global__ void flash_attn(const __bf16* __restrict__ q,
                           const __bf16* __restrict__ kk,
                           const __bf16* __restrict__ vT,
                           __bf16* __restrict__ attn_out) {
  int id = blockIdx.x;
  int mt = id % 144;
  int bh = id / 144;                       // b*HEADS + h
  int lane = threadIdx.x, li = lane & 15, half = lane >> 4;
  const float scale = 0.17677669529663687f;   // 1/sqrt(32)

  const __bf16* qrow = q + ((size_t)(bh * NPIX) + mt * 16 + li) * HD;
  bf16x16 qa = load_a(qrow, 0, half);        // fixed A fragment (K = 32 = d)

  f32x8 o0 = {}, o1 = {};
  float mrow[8], lrow[8];
#pragma unroll
  for (int r = 0; r < 8; ++r) { mrow[r] = -3.0e30f; lrow[r] = 0.0f; }

  __shared__ __bf16 plds[32 * 16];   // P tile, column-major: (m,k) at [k*16+m]

  for (int j0 = 0; j0 < NPIX; j0 += 32) {
    // S = Q K^T : B[dd][j] = k[j][dd] -> k rows are K-contiguous columns.
    bf16x16 kb0 = load_b(kk + ((size_t)(bh * NPIX) + j0 + li) * HD, 0, half);
    bf16x16 kb1 = load_b(kk + ((size_t)(bh * NPIX) + j0 + 16 + li) * HD, 0, half);
    f32x8 zc = {};
    f32x8 s0 = wmma_bf16f32(qa, kb0, zc);
    f32x8 s1 = wmma_bf16f32(qa, kb1, zc);

    float cmax[8];
#pragma unroll
    for (int r = 0; r < 8; ++r) {
      s0[r] *= scale; s1[r] *= scale;
      cmax[r] = fmaxf(s0[r], s1[r]);
    }
#pragma unroll
    for (int off = 1; off < 16; off <<= 1)
#pragma unroll
      for (int r = 0; r < 8; ++r)
        cmax[r] = fmaxf(cmax[r], __shfl_xor(cmax[r], off, 32));

    float fac[8], psum[8];
#pragma unroll
    for (int r = 0; r < 8; ++r) {
      float nm = fmaxf(mrow[r], cmax[r]);
      fac[r] = __expf(mrow[r] - nm);
      mrow[r] = nm;
      s0[r] = __expf(s0[r] - nm);
      s1[r] = __expf(s1[r] - nm);
      psum[r] = s0[r] + s1[r];
    }
#pragma unroll
    for (int off = 1; off < 16; off <<= 1)
#pragma unroll
      for (int r = 0; r < 8; ++r)
        psum[r] += __shfl_xor(psum[r], off, 32);
#pragma unroll
    for (int r = 0; r < 8; ++r) {
      lrow[r] = lrow[r] * fac[r] + psum[r];
      o0[r] *= fac[r];
      o1[r] *= fac[r];
    }

    // stage P in LDS (column-major) to convert C-layout -> A-layout
    __syncthreads();
#pragma unroll
    for (int r = 0; r < 8; ++r) {
      int m = half * 8 + r;
      plds[li * 16 + m]        = (__bf16)s0[r];     // columns j0+li
      plds[(16 + li) * 16 + m] = (__bf16)s1[r];     // columns j0+16+li
    }
    __syncthreads();

    BF16V pu;
#pragma unroll
    for (int t = 0; t < 8; ++t) {
      pu.e[t]     = plds[(half * 8 + t) * 16 + li];        // K = half*8 + t
      pu.e[8 + t] = plds[(16 + half * 8 + t) * 16 + li];   // K = 16+half*8+t
    }
    bf16x16 pa = pu.v;

    // O += P V : B[kk][dc] = vT[dc][j0+kk] -> contiguous in j.
    bf16x16 vb0 = load_b(vT + ((size_t)(bh * HD) + li) * NPIX + j0, 0, half);
    bf16x16 vb1 = load_b(vT + ((size_t)(bh * HD) + 16 + li) * NPIX + j0, 0, half);
    o0 = wmma_bf16f32(pa, vb0, o0);
    o1 = wmma_bf16f32(pa, vb1, o1);
  }

  int h = bh & (HEADS - 1), b = bh >> 3;
#pragma unroll
  for (int r = 0; r < 8; ++r) {
    int row = mt * 16 + half * 8 + r;
    float inv = 1.0f / lrow[r];
    size_t base = ((size_t)(b * NPIX) + row) * CH + h * HD;
    attn_out[base + li]      = (__bf16)(o0[r] * inv);
    attn_out[base + 16 + li] = (__bf16)(o1[r] * inv);
  }
}

// ---------------------------------------------------------------------------
// Kernel 5: output projection + residual.  src2 = src + attn_out @ wo^T + bo
// Writes f32 [b][C][NPIX] (channel-major, ready for conv + final residual).
// ---------------------------------------------------------------------------
__global__ void oproj_res(const __bf16* __restrict__ attn_out,
                          const __bf16* __restrict__ woB,   // [256][256] bf16
                          const float* __restrict__ bo,
                          const float* __restrict__ src,
                          float* __restrict__ src2) {
  int id = blockIdx.x;
  int ng = id & 3;   id >>= 2;
  int mt = id % 144;
  int b  = id / 144;
  int lane = threadIdx.x, li = lane & 15, half = lane >> 4;

  const __bf16* A = attn_out + ((size_t)(b * NPIX) + mt * 16 + li) * CH;
  int n0 = ng * 64;
  f32x8 acc[4] = {};
  for (int k0 = 0; k0 < CH; k0 += 32) {
    bf16x16 a = load_a(A, k0, half);
#pragma unroll
    for (int nt = 0; nt < 4; ++nt) {
      bf16x16 bf = load_b(woB + (size_t)(n0 + nt * 16 + li) * CH, k0, half);
      acc[nt] = wmma_bf16f32(a, bf, acc[nt]);
    }
  }
#pragma unroll
  for (int nt = 0; nt < 4; ++nt) {
    int col = n0 + nt * 16 + li;
    float bb = bo[col];
#pragma unroll
    for (int r = 0; r < 8; ++r) {
      int row = mt * 16 + half * 8 + r;
      size_t idx = ((size_t)(b * CH) + col) * NPIX + row;
      src2[idx] = src[idx] + acc[nt][r] + bb;
    }
  }
}

// ---------------------------------------------------------------------------
// Kernel 6: conv1 (3x3, 256 -> 1024) as 9 accumulated GEMM taps + GELU.
// A rows = shifted pixels (zero-padded border via per-lane predicated loads);
// B = w1p [tap][oc][ic] bf16, K-contiguous.  Output bf16 [b][n][1024].
// ---------------------------------------------------------------------------
__global__ void conv1_gelu(const __bf16* __restrict__ x2n,
                           const __bf16* __restrict__ w1p,
                           const float* __restrict__ b1,
                           __bf16* __restrict__ y1g) {
  int id = blockIdx.x;
  int ng = id % 16;  id /= 16;       // 16 groups of 64 over 1024 out channels
  int mt = id % 144;
  int b  = id / 144;
  int lane = threadIdx.x, li = lane & 15, half = lane >> 4;

  int prow = mt * 16 + li;
  int py = prow / 48, px = prow % 48;
  int n0 = ng * 64;
  f32x8 acc[4] = {};

  for (int t = 0; t < 9; ++t) {
    int dy = t / 3 - 1, dx = t % 3 - 1;
    int yy = py + dy, xx = px + dx;
    bool valid = (yy >= 0) & (yy < 48) & (xx >= 0) & (xx < 48);
    int sp = valid ? (yy * 48 + xx) : 0;
    const __bf16* A = x2n + ((size_t)(b * NPIX) + sp) * CH;
    const __bf16* W = w1p + (size_t)t * DCONV * CH;
    for (int k0 = 0; k0 < CH; k0 += 32) {
      bf16x16 a = valid ? load_a(A, k0, half) : zero16();
#pragma unroll
      for (int nt = 0; nt < 4; ++nt) {
        bf16x16 bf = load_b(W + (size_t)(n0 + nt * 16 + li) * CH, k0, half);
        acc[nt] = wmma_bf16f32(a, bf, acc[nt]);
      }
    }
  }
#pragma unroll
  for (int nt = 0; nt < 4; ++nt) {
    int col = n0 + nt * 16 + li;
    float bb = b1[col];
#pragma unroll
    for (int r = 0; r < 8; ++r) {
      int row = mt * 16 + half * 8 + r;
      float x = acc[nt][r] + bb;
      float g = 0.5f * x * (1.0f + erff(x * 0.70710678118654752f)); // exact GELU
      y1g[((size_t)(b * NPIX) + row) * DCONV + col] = (__bf16)g;
    }
  }
}

// ---------------------------------------------------------------------------
// Kernel 7: conv2 (1x1, 1024 -> 256) + final residual -> d_out (f32 NCHW).
// ---------------------------------------------------------------------------
__global__ void conv2_res(const __bf16* __restrict__ y1g,
                          const __bf16* __restrict__ w2p,   // [256][1024] bf16
                          const float* __restrict__ b2,
                          const float* __restrict__ src2,
                          float* __restrict__ out) {
  int id = blockIdx.x;
  int ng = id & 3;   id >>= 2;
  int mt = id % 144;
  int b  = id / 144;
  int lane = threadIdx.x, li = lane & 15, half = lane >> 4;

  const __bf16* A = y1g + ((size_t)(b * NPIX) + mt * 16 + li) * DCONV;
  int n0 = ng * 64;
  f32x8 acc[4] = {};
  for (int k0 = 0; k0 < DCONV; k0 += 32) {
    bf16x16 a = load_a(A, k0, half);
#pragma unroll
    for (int nt = 0; nt < 4; ++nt) {
      bf16x16 bf = load_b(w2p + (size_t)(n0 + nt * 16 + li) * DCONV, k0, half);
      acc[nt] = wmma_bf16f32(a, bf, acc[nt]);
    }
  }
#pragma unroll
  for (int nt = 0; nt < 4; ++nt) {
    int col = n0 + nt * 16 + li;
    float bb = b2[col];
#pragma unroll
    for (int r = 0; r < 8; ++r) {
      int row = mt * 16 + half * 8 + r;
      size_t idx = ((size_t)(b * CH) + col) * NPIX + row;
      out[idx] = src2[idx] + acc[nt][r] + bb;
    }
  }
}

// ---------------------------------------------------------------------------
// Host-side orchestration
// ---------------------------------------------------------------------------
extern "C" void kernel_launch(void* const* d_in, const int* in_sizes, int n_in,
                              void* d_out, int out_size, void* d_ws, size_t ws_size,
                              hipStream_t stream) {
  const float* src     = (const float*)d_in[0];
  const float* wq      = (const float*)d_in[1];
  const float* bq      = (const float*)d_in[2];
  const float* wk      = (const float*)d_in[3];
  const float* bk      = (const float*)d_in[4];
  const float* wv      = (const float*)d_in[5];
  const float* bv      = (const float*)d_in[6];
  const float* wo      = (const float*)d_in[7];
  const float* bo      = (const float*)d_in[8];
  const float* conv1_w = (const float*)d_in[9];
  const float* conv1_b = (const float*)d_in[10];
  const float* conv2_w = (const float*)d_in[11];
  const float* conv2_b = (const float*)d_in[12];
  const float* alpha1  = (const float*)d_in[13];
  const float* alpha2  = (const float*)d_in[14];
  float* outp = (float*)d_out;

  char* wsp = (char*)d_ws;
  auto alloc = [&](size_t bytes) -> void* {
    void* p = (void*)wsp;
    wsp += (bytes + 255) & ~(size_t)255;
    return p;
  };

  __bf16* xn       = (__bf16*)alloc((size_t)BATCH * NPIX * CH * 2);
  __bf16* wqkv     = (__bf16*)alloc((size_t)3 * CH * CH * 2);
  __bf16* woB      = (__bf16*)alloc((size_t)CH * CH * 2);
  __bf16* qb       = (__bf16*)alloc((size_t)BATCH * HEADS * NPIX * HD * 2);
  __bf16* kb       = (__bf16*)alloc((size_t)BATCH * HEADS * NPIX * HD * 2);
  __bf16* vTb      = (__bf16*)alloc((size_t)BATCH * HEADS * HD * NPIX * 2);
  __bf16* attn_out = (__bf16*)alloc((size_t)BATCH * NPIX * CH * 2);
  float*  src2     = (float*) alloc((size_t)BATCH * CH * NPIX * 4);
  __bf16* x2n      = (__bf16*)alloc((size_t)BATCH * NPIX * CH * 2);
  __bf16* w1p      = (__bf16*)alloc((size_t)9 * DCONV * CH * 2);
  __bf16* w2p      = (__bf16*)alloc((size_t)CH * DCONV * 2);
  __bf16* y1g      = (__bf16*)alloc((size_t)BATCH * NPIX * DCONV * 2);

  // 1) first pixel norm + bf16 pack
  pixelnorm_pack<<<BATCH * NPIX, CH, 0, stream>>>(src, alpha1, xn);

  // 2) weight packing
  f32_to_bf16<<<(CH * CH + 255) / 256, 256, 0, stream>>>(wq, wqkv + 0 * CH * CH, CH * CH);
  f32_to_bf16<<<(CH * CH + 255) / 256, 256, 0, stream>>>(wk, wqkv + 1 * CH * CH, CH * CH);
  f32_to_bf16<<<(CH * CH + 255) / 256, 256, 0, stream>>>(wv, wqkv + 2 * CH * CH, CH * CH);
  f32_to_bf16<<<(CH * CH + 255) / 256, 256, 0, stream>>>(wo, woB, CH * CH);
  f32_to_bf16<<<(CH * DCONV + 255) / 256, 256, 0, stream>>>(conv2_w, w2p, CH * DCONV);
  pack_conv1<<<(DCONV * CH * 9 + 255) / 256, 256, 0, stream>>>(conv1_w, w1p);

  // 3) QKV projections (WMMA)
  qkv_gemm<<<3 * BATCH * 144 * 4, 32, 0, stream>>>(xn, wqkv, bq, bk, bv, qb, kb, vTb);

  // 4) streaming attention (WMMA)
  flash_attn<<<BATCH * HEADS * 144, 32, 0, stream>>>(qb, kb, vTb, attn_out);

  // 5) output projection + residual (WMMA)
  oproj_res<<<BATCH * 144 * 4, 32, 0, stream>>>(attn_out, woB, bo, src, src2);

  // 6) second pixel norm + bf16 pack
  pixelnorm_pack<<<BATCH * NPIX, CH, 0, stream>>>(src2, alpha2, x2n);

  // 7) 3x3 conv as 9 GEMM taps + GELU (WMMA)
  conv1_gelu<<<BATCH * 144 * 16, 32, 0, stream>>>(x2n, w1p, conv1_b, y1g);

  // 8) 1x1 conv + final residual -> output (WMMA)
  conv2_res<<<BATCH * 144 * 4, 32, 0, stream>>>(y1g, w2p, conv2_b, src2, outp);
}
